// Self_Attn_36661840838720
// MI455X (gfx1250) — matrix operand order repaired and checked by reading, not compiled, not run on hardware
//
#include <hip/hip_runtime.h>
#include <hip/hip_bf16.h>

typedef __attribute__((ext_vector_type(16))) __bf16 v16bf;
typedef __attribute__((ext_vector_type(8)))  __bf16 v8bf;
typedef __attribute__((ext_vector_type(2)))  __bf16 v2bf;
typedef __attribute__((ext_vector_type(8)))  float  v8f;
typedef __attribute__((ext_vector_type(4)))  float  v4f;

#define NTOK 169   // 13*13 tokens
#define NP   176   // padded to 11*16
#define KPF  192   // final-GEMM K padded to 6*32
#define LDX  520   // Xcm k-stride (512+8): conflict-free, 16B aligned
#define LDP  72    // PFT/PGcm k-stride (64+8)
#define LDF8 264   // F8cm k-stride (256+8)
#define LDZ  136   // Zcm  k-stride (128+8)
#define LDMB 200   // Mbcm k-stride (192+8)

// ---------------- WMMA fragment helpers (CDNA5 16x16x32 bf16 layouts) -------
// A 16x32: lane m=lane&15; k runs {half*8+[0,8)} and {16+half*8+[0,8)} -> 2x16B
// B 32x16: lane n=lane&15; k runs {half*8+[0,8)} and {16+half*8+[0,8)} (column-major) -> 2x16B
// C 16x16: row = r + 8*half, col = lane&15

__device__ __forceinline__ v16bf cat8(v8bf lo, v8bf hi) {
  v16bf r;
#pragma unroll
  for (int i = 0; i < 8; ++i) { r[i] = lo[i]; r[i + 8] = hi[i]; }
  return r;
}

// A fragment from row-major bf16 LDS tile (lda elements, 16B-aligned rows)
__device__ __forceinline__ v16bf lds_load_a_rm(const __bf16* A, int lda, int lane) {
  const int half = lane >> 4, m = lane & 15;
  const __bf16* base = A + m * lda + (half << 3);
  return cat8(*(const v8bf*)(base), *(const v8bf*)(base + 16));
}

// B fragment from column-major bf16 LDS panel: Bcm[col][k], ldk elements
__device__ __forceinline__ v16bf lds_load_b_cm(const __bf16* Bcm, int ldk,
                                               int kb_base, int n0, int lane) {
  const int n  = n0 + (lane & 15);
  const int kb = kb_base + ((lane >> 4) << 3);
  const __bf16* p = Bcm + n * ldk + kb;
  return cat8(*(const v8bf*)(p), *(const v8bf*)(p + 16));
}

// A fragment from global f32 (vectorized; requires lda % 4 == 0), row-bounded
__device__ __forceinline__ v16bf glb_load_a_f32v(const float* __restrict__ A, int lda,
                                                 int m0, int k0, int mmax, int lane) {
  const int half = lane >> 4;
  const int m = m0 + (lane & 15);
  v4f r[4] = {{}, {}, {}, {}};
  if (m < mmax) {
    const float* base = A + m * lda + k0 + (half << 3);
    r[0] = *(const v4f*)(base);
    r[1] = *(const v4f*)(base + 4);
    r[2] = *(const v4f*)(base + 16);
    r[3] = *(const v4f*)(base + 20);
  }
  v16bf a;
#pragma unroll
  for (int i = 0; i < 16; ++i) a[i] = (__bf16)r[i >> 2][i & 3];
  return a;
}

// A fragment from global f32, scalar + k-bounded (for c_w: lda=169, unaligned)
__device__ __forceinline__ v16bf glb_load_a_f32s(const float* __restrict__ A, int lda,
                                                 int m0, int k0, int kmax, int lane) {
  const int half = lane >> 4;
  const int m = m0 + (lane & 15);
  v16bf a;
#pragma unroll
  for (int i = 0; i < 16; ++i) {
    const int k = k0 + ((i >> 3) << 4) + (half << 3) + (i & 7);
    a[i] = (__bf16)((k < kmax) ? A[m * lda + k] : 0.0f);
  }
  return a;
}

__device__ __forceinline__ v8f wmma_bf16(v16bf a, v16bf b, v8f c) {
  return __builtin_amdgcn_wmma_f32_16x16x32_bf16(false, a, false, b, (short)0, c,
                                                 false, false);
}

// W[169 x K] (global f32, lda=K multiple of 32) @ Bcm[176 cols][K] -> += Mf (+bias)
__device__ __forceinline__ void gemm_add_phase(const float* __restrict__ W,
                                               const float* __restrict__ bias, int K,
                                               const __bf16* Bcm, int ldk, float* Mf,
                                               int w, int lane) {
  const int half = lane >> 4, nl = lane & 15;
  const int nkc = K >> 5;
  for (int t = w; t < 121; t += 8) {
    const int tm = t / 11, tn = t - tm * 11;
    v8f acc = {};
    for (int kc = 0; kc < nkc; ++kc) {
      v16bf a = glb_load_a_f32v(W, K, tm * 16, kc * 32, NTOK, lane);
      v16bf b = lds_load_b_cm(Bcm, ldk, kc * 32, tn * 16, lane);
      acc = wmma_bf16(a, b, acc);
    }
#pragma unroll
    for (int r = 0; r < 8; ++r) {
      const int row = tm * 16 + r + (half << 3);
      const int col = tn * 16 + nl;
      const float bv = (row < NTOK) ? bias[row] : 0.0f;
      Mf[row * NP + col] += acc[r] + bv;
    }
  }
}

// ---------------------------------------------------------------------------

__global__ void __launch_bounds__(256, 1)
self_attn_fused(const float* __restrict__ x,  const float* __restrict__ f5,
                const float* __restrict__ f8, const float* __restrict__ f_w,
                const float* __restrict__ f_b, const float* __restrict__ g_w,
                const float* __restrict__ g_b, const float* __restrict__ h_w,
                const float* __restrict__ h_b, const float* __restrict__ u_w,
                const float* __restrict__ u_b, const float* __restrict__ c_w,
                const float* __restrict__ c_b, const float* __restrict__ gammap,
                float* __restrict__ out) {
  extern __shared__ char smem[];
  // region plan (bytes):
  //  [0,183040)        Xcm  bf16 [176][520]  (phase 1)
  //  [183040,208384)   PFT  bf16 [176][72]   (phase 1 -> 2)   PF^T: [n][c]
  //  [208384,233728)   PGcm bf16 [176][72]   (phase 1 -> 2)   PG col-major: [n][c]
  //  [0,123904)        Mf   f32  [176][176]  (phase 2..6, overlaps Xcm)
  //  [123904,216832)   F8cm bf16 [176][264]  (phase 4)
  //  [123904,171776)   Zcm  bf16 [176][136]  (phase 5)
  //  [123904,194304)   Mbcm bf16 [176][200]  (phase 6)
  __bf16* Xcm  = (__bf16*)(smem);
  __bf16* PFT  = (__bf16*)(smem + 183040);
  __bf16* PGcm = (__bf16*)(smem + 208384);
  float*  Mf   = (float*)(smem);
  __bf16* F8cm = (__bf16*)(smem + 123904);
  __bf16* Zcm  = (__bf16*)(smem + 123904);
  __bf16* Mbcm = (__bf16*)(smem + 123904);

  const int b    = blockIdx.x;
  const int tid  = threadIdx.x;
  const int w    = tid >> 5;
  const int lane = tid & 31;
  const int half = lane >> 4, nl = lane & 15;

  const float* xb  = x  + (size_t)b * 512 * NTOK;
  const float* f5b = f5 + (size_t)b * 128 * 676;
  const float* f8p = f8 + (size_t)b * 256 * NTOK;
  float*       ob  = out + (size_t)b * 512 * NTOK;

  // ---- Phase 1a: X [512x169] f32 -> Xcm[col][k] bf16, packed b32 stores ----
  for (int idx = tid; idx < 256 * NP; idx += 256) {
    const int c2 = idx / NP, s = idx - c2 * NP;
    const int c = c2 * 2;
    v2bf t = {};
    if (s < NTOK) {
      t[0] = (__bf16)xb[c * NTOK + s];
      t[1] = (__bf16)xb[(c + 1) * NTOK + s];
    }
    *(v2bf*)(Xcm + s * LDX + c) = t;
  }
  __syncthreads();

  // ---- Phase 1b: PF = f_w@X+f_b (waves 0-3), PG = g_w@X+g_b (waves 4-7) ----
  // Both results stored as [n][c] (ld=72): PF^T is A of energy, PGcm is B of energy.
  {
    const float* W   = (w < 4) ? f_w : g_w;
    const float* Wb  = (w < 4) ? f_b : g_b;
    __bf16* dstT     = (w < 4) ? PFT : PGcm;   // same address math both sides
    const int rt = w & 3;
    for (int tn = 0; tn < 11; ++tn) {
      v8f acc = {};
      for (int kc = 0; kc < 16; ++kc) {
        v16bf a  = glb_load_a_f32v(W, 512, rt * 16, kc * 32, 64, lane);
        v16bf bb = lds_load_b_cm(Xcm, LDX, kc * 32, tn * 16, lane);
        acc = wmma_bf16(a, bb, acc);
      }
#pragma unroll
      for (int r = 0; r < 8; ++r) {
        const int m = rt * 16 + r + (half << 3);
        const int n = tn * 16 + nl;
        dstT[n * LDP + m] = (__bf16)(acc[r] + Wb[m]);
      }
    }
  }
  __syncthreads();

  // ---- Phase 2: energy[n,m] = sum_c PFT[n][c]*PGcm[m][c]  ->  Mf f32 ----
  for (int t = w; t < 121; t += 8) {
    const int tm = t / 11, tn = t - tm * 11;
    v8f acc = {};
#pragma unroll
    for (int kc = 0; kc < 2; ++kc) {
      v16bf a  = lds_load_a_rm(PFT + (tm * 16) * LDP + kc * 32, LDP, lane);
      v16bf bb = lds_load_b_cm(PGcm, LDP, kc * 32, tn * 16, lane);
      acc = wmma_bf16(a, bb, acc);
    }
#pragma unroll
    for (int r = 0; r < 8; ++r)
      Mf[(tm * 16 + r + (half << 3)) * NP + tn * 16 + nl] = acc[r];
  }
  __syncthreads();

  // ---- Phase 3: row softmax over the 169 valid columns ----
  if (tid < NTOK) {
    float* row = Mf + tid * NP;
    float mx = row[0];
    for (int s = 1; s < NTOK; ++s) mx = fmaxf(mx, row[s]);
    float sum = 0.0f;
    for (int s = 0; s < NTOK; ++s) { const float e = __expf(row[s] - mx); row[s] = e; sum += e; }
    const float inv = 1.0f / sum;
    for (int s = 0; s < NTOK; ++s) row[s] *= inv;
  }
  __syncthreads();

  // ---- Phase 4: += h_w @ F8 + h_b ----
  for (int idx = tid; idx < 128 * NP; idx += 256) {
    const int c2 = idx / NP, s = idx - c2 * NP;
    const int c = c2 * 2;
    v2bf t = {};
    if (s < NTOK) {
      t[0] = (__bf16)f8p[c * NTOK + s];
      t[1] = (__bf16)f8p[(c + 1) * NTOK + s];
    }
    *(v2bf*)(F8cm + s * LDF8 + c) = t;
  }
  __syncthreads();
  gemm_add_phase(h_w, h_b, 256, F8cm, LDF8, Mf, w, lane);
  __syncthreads();

  // ---- Phase 5: Z = bilinear(26->13, half-pixel) == 2x2 avg pool; += u_w@Z + u_b
  for (int idx = tid; idx < 64 * NP; idx += 256) {
    const int c2 = idx / NP, s = idx - c2 * NP;
    const int c = c2 * 2;
    v2bf t = {};
    if (s < NTOK) {
      const int i = s / 13, j = s - i * 13;
      const float* p0 = f5b + c * 676 + (2 * i) * 26 + 2 * j;
      const float* p1 = p0 + 676;
      t[0] = (__bf16)(0.25f * (p0[0] + p0[1] + p0[26] + p0[27]));
      t[1] = (__bf16)(0.25f * (p1[0] + p1[1] + p1[26] + p1[27]));
    }
    *(v2bf*)(Zcm + s * LDZ + c) = t;
  }
  __syncthreads();
  gemm_add_phase(u_w, u_b, 128, Zcm, LDZ, Mf, w, lane);
  __syncthreads();

  // ---- Phase 6a: M f32 -> Mbcm[col][k] bf16, pad rows zeroed (NaN-safe) ----
  for (int idx = tid; idx < NP * (KPF / 2); idx += 256) {
    const int s = idx / (KPF / 2), r2 = idx - s * (KPF / 2);
    const int row = r2 * 2;
    v2bf t;
    t[0] = (row     < NTOK) ? (__bf16)Mf[row * NP + s]       : (__bf16)0.0f;
    t[1] = (row + 1 < NTOK) ? (__bf16)Mf[(row + 1) * NP + s] : (__bf16)0.0f;
    *(v2bf*)(Mbcm + s * LDMB + row) = t;
  }
  __syncthreads();

  // ---- Phase 6b: out = gamma*(c_w@M + c_b) + x ----
  const float gamma = gammap[0];
  for (int t = w; t < 352; t += 8) {           // 32 row tiles x 11 col tiles
    const int tm = t / 11, tn = t - tm * 11;
    v8f acc = {};
#pragma unroll
    for (int kc = 0; kc < 6; ++kc) {
      v16bf a  = glb_load_a_f32s(c_w, NTOK, tm * 16, kc * 32, NTOK, lane);
      v16bf bb = lds_load_b_cm(Mbcm, LDMB, kc * 32, tn * 16, lane);
      acc = wmma_bf16(a, bb, acc);
    }
#pragma unroll
    for (int r = 0; r < 8; ++r) {
      const int o = tm * 16 + r + (half << 3);
      const int s = tn * 16 + nl;
      if (s < NTOK) {
        ob[o * NTOK + s] = gamma * (acc[r] + c_b[o]) + xb[o * NTOK + s];
      }
    }
  }
}

// ---------------------------------------------------------------------------

extern "C" void kernel_launch(void* const* d_in, const int* in_sizes, int n_in,
                              void* d_out, int out_size, void* d_ws, size_t ws_size,
                              hipStream_t stream) {
  const float* x   = (const float*)d_in[0];
  const float* f5  = (const float*)d_in[1];
  const float* f8  = (const float*)d_in[2];
  const float* f_w = (const float*)d_in[3];
  const float* f_b = (const float*)d_in[4];
  const float* g_w = (const float*)d_in[5];
  const float* g_b = (const float*)d_in[6];
  const float* h_w = (const float*)d_in[7];
  const float* h_b = (const float*)d_in[8];
  const float* u_w = (const float*)d_in[9];
  const float* u_b = (const float*)d_in[10];
  const float* c_w = (const float*)d_in[11];
  const float* c_b = (const float*)d_in[12];
  const float* gm  = (const float*)d_in[13];
  float* out = (float*)d_out;

  const int B = in_sizes[0] / (512 * NTOK);
  const size_t smem = 233728;  // see region plan in kernel

  (void)hipFuncSetAttribute((const void*)self_attn_fused,
                            hipFuncAttributeMaxDynamicSharedMemorySize, (int)smem);

  self_attn_fused<<<dim3(B), dim3(256), smem, stream>>>(
      x, f5, f8, f_w, f_b, g_w, g_b, h_w, h_b, u_w, u_b, c_w, c_b, gm, out);
  (void)d_ws; (void)ws_size; (void)n_in; (void)out_size;
}
